// MHAwithKVCache_38405597561414
// MI455X (gfx1250) — matrix operand order repaired and checked by reading, not compiled, not run on hardware
//
#include <hip/hip_runtime.h>

// ---------------------------------------------------------------------------
// CDNA5 (gfx1250, wave32) MHA forward: bf16 WMMA + TDM async tensor staging.
// ---------------------------------------------------------------------------

typedef __bf16 bf16_t;
typedef __attribute__((ext_vector_type(16))) __bf16 v16bf;
typedef __attribute__((ext_vector_type(8)))  __bf16 v8bf;
typedef __attribute__((ext_vector_type(8)))  float  v8f;
typedef __attribute__((ext_vector_type(4)))  unsigned v4u;
typedef __attribute__((ext_vector_type(8)))  int      v8i_;
typedef __attribute__((ext_vector_type(4)))  int      v4i_;

#define WMMA_BF16(a, b, c) \
  __builtin_amdgcn_wmma_f32_16x16x32_bf16(false, (a), false, (b), (short)0, (c), false, false)

#if __has_builtin(__builtin_amdgcn_tensor_load_to_lds) && \
    __has_builtin(__builtin_amdgcn_s_wait_tensorcnt)
#define HAVE_TDM 1
#else
#define HAVE_TDM 0
#endif

static constexpr int Bn = 4;
static constexpr int Hn = 16;
static constexpr int Sn = 2048;
static constexpr int DK = 64;
static constexpr int Dm = 1024;     // D_MODEL
static constexpr int Mn = Bn * Sn;  // 8192 rows in the projection GEMMs

// ---- fp32 -> bf16, round to nearest even -----------------------------------
__device__ inline bf16_t f2bf(float f) {
  union { float f; unsigned u; } in;
  in.f = f;
  unsigned u = in.u;
  u += 0x7FFFu + ((u >> 16) & 1u);
  union { unsigned short s; bf16_t b; } out;
  out.s = (unsigned short)(u >> 16);
  return out.b;
}

// ---- operand assembly helpers (ISA §7.12.2 VGPR layouts) -------------------
__device__ inline v16bf cat16(v8bf lo, v8bf hi) {
  v16bf r;
#pragma unroll
  for (int i = 0; i < 8; i++) { r[i] = lo[i]; r[8 + i] = hi[i]; }
  return r;
}

// B operand: 16 contiguous bf16 at p (32B aligned): lane n=l%16, elem i<->K=16h+i
__device__ inline v16bf loadv16(const bf16_t* p) {
  return *(const v16bf*)p;
}

// A operand, bf16 row-major. rowp = &M[row m][k0]; h = lane>>4.
// elem i <-> K = 16h+i
__device__ inline v16bf load_A_bf(const bf16_t* rowp, int h) {
  return cat16(*(const v8bf*)(rowp + 16 * h), *(const v8bf*)(rowp + 16 * h + 8));
}

// A operand, fp32 row-major source, converted on the fly.
__device__ inline v16bf load_A_f32(const float* rowp, int h) {
  v16bf a;
#pragma unroll
  for (int i = 0; i < 8; i++) {
    a[i]     = f2bf(rowp[16 * h + i]);
    a[8 + i] = f2bf(rowp[16 * h + 8 + i]);
  }
  return a;
}

// ---- Tensor Data Mover: 2D tile (rows x row_len bf16) global -> LDS --------
// D# per ISA 08_async_tensor §8: group0 = {flags, lds_addr, global_addr, type=2},
// group1 = {data_size=2B, tensor dims, tile dims, dim0 stride}.
// amdgpu-toolchain (clang-23) 6-arg builtin: (v4u, v8i, v4i, v4i, v8i, i32 cpol)
__device__ inline void tdm_load_2d(unsigned lds_off, const void* gaddr,
                                   unsigned tile_d0, unsigned tile_d1,
                                   unsigned tensor_d0, unsigned tensor_d1,
                                   unsigned stride0_elems) {
#if HAVE_TDM
  const unsigned long long ga = (unsigned long long)(uintptr_t)gaddr;
  v4u g0;
  g0[0] = 1u;                                        // count=1, user mode
  g0[1] = lds_off;                                   // lds_addr (bytes)
  g0[2] = (unsigned)(ga & 0xFFFFFFFFu);              // global_addr[31:0]
  g0[3] = (unsigned)((ga >> 32) & 0x01FFFFFFu)       // global_addr[56:32]
          | 0x80000000u;                             // type=2 ("image")
  v8i_ g1;
  g1[0] = (int)(1u << 16);                           // data_size=1 -> 2 bytes
  g1[1] = (int)(tensor_d0 << 16);                    // tensor_dim0[15:0]
  g1[2] = (int)((tensor_d0 >> 16) | (tensor_d1 << 16)); // dim0 hi | dim1 lo
  g1[3] = (int)(((tensor_d1 >> 16) & 0xFFFFu) | (tile_d0 << 16)); // dim1 hi | tile_dim0
  g1[4] = (int)(tile_d1 & 0xFFFFu);                  // tile_dim1 (tile_dim2=0)
  g1[5] = (int)stride0_elems;                        // tensor_dim0_stride lo
  g1[6] = 0;                                         // stride hi / dim1_stride
  g1[7] = 0;
  v4i_ z4 = {};
  v8i_ z8 = {};
  __builtin_amdgcn_tensor_load_to_lds(g0, g1, z4, z4, z8, 0);
#endif
}

// ---------------------------------------------------------------------------
// Weight conversion fp32 -> bf16
// ---------------------------------------------------------------------------
__global__ __launch_bounds__(256) void cvt_f32_bf16(const float* __restrict__ in,
                                                    bf16_t* __restrict__ out, int n) {
  int i = blockIdx.x * blockDim.x + threadIdx.x;
  if (i < n) out[i] = f2bf(in[i]);
}

// ---------------------------------------------------------------------------
// GEMM: Y = A @ W^T (M=8192, N=1024, K=1024), one wave -> 32x64 output tile.
// MODE 0: A fp32 (q),  write bf16 [B,H,S,64]
// MODE 1: A fp32 (k),  write fp32 key [B,H,S,64] + bf16 same
// MODE 2: A fp32 (v),  write fp32 value [B,H,S,64] + bf16 [B,H,64,S] (V^T)
// MODE 3: A bf16 (ctx),write fp32 [B,S,1024]
// ---------------------------------------------------------------------------
template <int MODE>
__global__ __launch_bounds__(32) void gemm_kernel(const void* __restrict__ Ap,
                                                  const bf16_t* __restrict__ Wb,
                                                  float* __restrict__ f32out,
                                                  bf16_t* __restrict__ bf16out) {
  const int lane = threadIdx.x & 31;
  const int nl = lane & 15, hh = lane >> 4;
  const int m0 = blockIdx.x * 32;
  const int n0 = blockIdx.y * 64;

  v8f acc[2][4];
#pragma unroll
  for (int mi = 0; mi < 2; mi++)
#pragma unroll
    for (int t = 0; t < 4; t++) acc[mi][t] = v8f{};

  for (int k0 = 0; k0 < Dm; k0 += 32) {
    v16bf a0, a1;
    if constexpr (MODE == 3) {
      const bf16_t* A = (const bf16_t*)Ap;
      a0 = load_A_bf(A + (size_t)(m0 + nl) * Dm + k0, hh);
      a1 = load_A_bf(A + (size_t)(m0 + 16 + nl) * Dm + k0, hh);
      if (k0 + 32 < Dm) {
        __builtin_prefetch(A + (size_t)(m0 + nl) * Dm + k0 + 32, 0, 3);
        __builtin_prefetch(A + (size_t)(m0 + 16 + nl) * Dm + k0 + 32, 0, 3);
      }
    } else {
      const float* A = (const float*)Ap;
      a0 = load_A_f32(A + (size_t)(m0 + nl) * Dm + k0, hh);
      a1 = load_A_f32(A + (size_t)(m0 + 16 + nl) * Dm + k0, hh);
      if (k0 + 32 < Dm) {
        __builtin_prefetch(A + (size_t)(m0 + nl) * Dm + k0 + 32, 0, 3);
        __builtin_prefetch(A + (size_t)(m0 + 16 + nl) * Dm + k0 + 32, 0, 3);
      }
    }
#pragma unroll
    for (int t = 0; t < 4; t++) {
      // B[k][n] = W[n][k]: row n of W, contiguous 16 bf16 at k0+16h (32B aligned)
      v16bf bw = loadv16(Wb + (size_t)(n0 + 16 * t + nl) * Dm + k0 + 16 * hh);
      acc[0][t] = WMMA_BF16(a0, bw, acc[0][t]);
      acc[1][t] = WMMA_BF16(a1, bw, acc[1][t]);
    }
  }

#pragma unroll
  for (int mi = 0; mi < 2; mi++) {
#pragma unroll
    for (int t = 0; t < 4; t++) {
#pragma unroll
      for (int r = 0; r < 8; r++) {
        const int m  = m0 + 16 * mi + r + 8 * hh;  // C/D: row = r + 8*(l/16)
        const int nn = n0 + 16 * t + nl;           //      col = l%16
        const float val = acc[mi][t][r];
        if constexpr (MODE == 3) {
          f32out[(size_t)m * Dm + nn] = val;
        } else {
          const int bb = m >> 11, s = m & (Sn - 1);
          const int hd = nn >> 6, d = nn & (DK - 1);
          const size_t hidx = ((size_t)(bb * Hn + hd) * Sn + s) * DK + d;
          if constexpr (MODE == 1 || MODE == 2) f32out[hidx] = val;
          if constexpr (MODE == 2)
            bf16out[((size_t)(bb * Hn + hd) * DK + d) * Sn + s] = f2bf(val);  // V^T
          else
            bf16out[hidx] = f2bf(val);
        }
      }
    }
  }
}

// ---------------------------------------------------------------------------
// Flash attention: one wave per (b,h, 16-query tile). Online softmax, causal.
// K/V tiles staged into LDS by the Tensor Data Mover, double-buffered.
// Q: bf16 [B,H,S,64], K: bf16 [B,H,S,64], Vt: bf16 [B,H,64,S],
// Ctx out: bf16 [B,S,H*64] for the final projection.
// ---------------------------------------------------------------------------
__global__ __launch_bounds__(32) void attn_kernel(const bf16_t* __restrict__ Qb,
                                                  const bf16_t* __restrict__ Kb,
                                                  const bf16_t* __restrict__ Vt,
                                                  bf16_t* __restrict__ Ctx) {
#if HAVE_TDM
  __shared__ __align__(32) bf16_t kLds[2][32 * DK];  // 2 x 4KB K tiles
  __shared__ __align__(32) bf16_t vLds[2][DK * 32];  // 2 x 4KB V^T tiles
#endif
  __shared__ __align__(16) bf16_t pLds[16 * 32];     // P tile (D->A relayout)

  const int lane = threadIdx.x & 31;
  const int nl = lane & 15, hh = lane >> 4;
  const int qt = blockIdx.x;  // query tile 0..127
  const int bh = blockIdx.y;  // b*16+h
  const int b = bh >> 4, h = bh & 15;

  const bf16_t* Qtile = Qb + ((size_t)bh * Sn + qt * 16) * DK;
  const bf16_t* qrow = Qtile + (size_t)nl * DK;
  const v16bf aq0 = load_A_bf(qrow, hh);
  const v16bf aq1 = load_A_bf(qrow + 32, hh);

  float mrow[8], lrow[8];
  v8f o[4];
#pragma unroll
  for (int r = 0; r < 8; r++) { mrow[r] = -1e30f; lrow[r] = 0.f; }
#pragma unroll
  for (int t = 0; t < 4; t++) o[t] = v8f{};

  const bf16_t* Kbase = Kb + (size_t)bh * Sn * DK;
  const bf16_t* Vbase = Vt + (size_t)bh * DK * Sn;
  const int nch = (16 * (qt + 1) + 31) >> 5;  // 32-key chunks (causal bound)

#if HAVE_TDM
  // Prologue: DMA chunk 0's K (32x64) and V^T (64x32) tiles into LDS buffer 0.
  tdm_load_2d((unsigned)(uintptr_t)&kLds[0][0], Kbase, DK, 32, DK, Sn, DK);
  tdm_load_2d((unsigned)(uintptr_t)&vLds[0][0], Vbase, 32, DK, 32, DK, Sn);
#endif

  for (int c = 0; c < nch; c++) {
    const int k0 = c << 5;
#if HAVE_TDM
    if (c + 1 < nch) {  // issue next chunk's tiles into the other buffer
      const int kn = (c + 1) << 5;
      tdm_load_2d((unsigned)(uintptr_t)&kLds[(c + 1) & 1][0],
                  Kbase + (size_t)kn * DK, DK, 32, DK, Sn - kn, DK);
      tdm_load_2d((unsigned)(uintptr_t)&vLds[(c + 1) & 1][0],
                  Vbase + kn, 32, DK, 32, DK, Sn);
      __builtin_amdgcn_s_wait_tensorcnt(2);  // chunk c's two tiles done
    } else {
      __builtin_amdgcn_s_wait_tensorcnt(0);
    }
    const bf16_t* kc = &kLds[c & 1][0];  // rows = local key, ld = 64
    const bf16_t* vc = &vLds[c & 1][0];  // rows = dk,        ld = 32
    const int vld = 32;
#else
    const bf16_t* kc = Kbase + (size_t)k0 * DK;
    const bf16_t* vc = Vbase + k0;
    const int vld = Sn;
#endif

    float sc[2][8], rm[8];
#pragma unroll
    for (int r = 0; r < 8; r++) rm[r] = -1e30f;

#pragma unroll
    for (int u = 0; u < 2; u++) {
      // B[k][n] = K[key n][dk k]: local row (16u+nl), contiguous chunks
      const bf16_t* krow = kc + (size_t)(16 * u + nl) * DK;
      v16bf bk0 = loadv16(krow + 16 * hh);
      v16bf bk1 = loadv16(krow + 32 + 16 * hh);
      v8f s = v8f{};
      s = WMMA_BF16(aq0, bk0, s);
      s = WMMA_BF16(aq1, bk1, s);
#pragma unroll
      for (int r = 0; r < 8; r++) {
        const int row = qt * 16 + r + 8 * hh;
        const int col = k0 + 16 * u + nl;
        float val = s[r] * 0.125f;   // 1/sqrt(64)
        if (col > row) val = -1e30f; // causal mask
        sc[u][r] = val;
        rm[r] = fmaxf(rm[r], val);
      }
    }
    // row max across the 16 lanes holding one row (xor<=8 stays in half-wave)
#pragma unroll
    for (int r = 0; r < 8; r++) {
      float v = rm[r];
      v = fmaxf(v, __shfl_xor(v, 1));
      v = fmaxf(v, __shfl_xor(v, 2));
      v = fmaxf(v, __shfl_xor(v, 4));
      v = fmaxf(v, __shfl_xor(v, 8));
      rm[r] = v;
    }
    float alpha[8], p0[8], p1[8];
#pragma unroll
    for (int r = 0; r < 8; r++) {
      const float mn = fmaxf(mrow[r], rm[r]);
      alpha[r] = __expf(mrow[r] - mn);
      mrow[r] = mn;
      p0[r] = __expf(sc[0][r] - mn);
      p1[r] = __expf(sc[1][r] - mn);
      float rs = p0[r] + p1[r];
      rs += __shfl_xor(rs, 1);
      rs += __shfl_xor(rs, 2);
      rs += __shfl_xor(rs, 4);
      rs += __shfl_xor(rs, 8);
      lrow[r] = lrow[r] * alpha[r] + rs;
    }
#pragma unroll
    for (int t = 0; t < 4; t++)
#pragma unroll
      for (int r = 0; r < 8; r++) o[t][r] *= alpha[r];

    // D-layout -> A-layout for P via LDS (16 rows x 32 cols, bf16)
#pragma unroll
    for (int r = 0; r < 8; r++) {
      pLds[(r + 8 * hh) * 32 + nl]      = f2bf(p0[r]);
      pLds[(r + 8 * hh) * 32 + 16 + nl] = f2bf(p1[r]);
    }
    __syncthreads();
    const v16bf ap = load_A_bf(&pLds[nl * 32], hh);

    // PV: B[k][n] = V[key k0+k][dk 16t+n] = Vt row (16t+n), contiguous keys
#pragma unroll
    for (int t = 0; t < 4; t++) {
      v16bf bv = loadv16(vc + (size_t)(16 * t + nl) * vld + 16 * hh);
      o[t] = WMMA_BF16(ap, bv, o[t]);
    }
    __syncthreads();
  }

  // normalize and store context in [B, S, H, 64] (== [B,S,1024])
#pragma unroll
  for (int r = 0; r < 8; r++) {
    const float inv = 1.f / lrow[r];
    const int sg = qt * 16 + r + 8 * hh;
    bf16_t* crow = Ctx + ((size_t)(b * Sn + sg) * Hn + h) * DK;
#pragma unroll
    for (int t = 0; t < 4; t++) crow[16 * t + nl] = f2bf(o[t][r] * inv);
  }
}

// ---------------------------------------------------------------------------
extern "C" void kernel_launch(void* const* d_in, const int* in_sizes, int n_in,
                              void* d_out, int out_size, void* d_ws, size_t ws_size,
                              hipStream_t stream) {
  const float* q  = (const float*)d_in[0];
  const float* k  = (const float*)d_in[1];
  const float* v  = (const float*)d_in[2];
  const float* Wq = (const float*)d_in[3];
  const float* Wk = (const float*)d_in[4];
  const float* Wv = (const float*)d_in[5];
  const float* Wo = (const float*)d_in[6];

  float* out  = (float*)d_out;                     // [B,S,1024]
  float* keyo = out + (size_t)Bn * Sn * Dm;        // [B,H,S,64]
  float* valo = keyo + (size_t)Bn * Hn * Sn * DK;  // [B,H,S,64]

  // Workspace (bf16): 4 weights (8 MB) + Q/K/V^T (48 MB) + Ctx (16 MB)
  bf16_t* Wqb = (bf16_t*)d_ws;
  bf16_t* Wkb = Wqb + (size_t)Dm * Dm;
  bf16_t* Wvb = Wkb + (size_t)Dm * Dm;
  bf16_t* Wob = Wvb + (size_t)Dm * Dm;
  bf16_t* Qbf = Wob + (size_t)Dm * Dm;
  bf16_t* Kbf = Qbf + (size_t)Mn * Dm;
  bf16_t* Vtb = Kbf + (size_t)Mn * Dm;
  bf16_t* Ctx = Vtb + (size_t)Mn * Dm;

  const int wn = Dm * Dm;
  cvt_f32_bf16<<<wn / 256, 256, 0, stream>>>(Wq, Wqb, wn);
  cvt_f32_bf16<<<wn / 256, 256, 0, stream>>>(Wk, Wkb, wn);
  cvt_f32_bf16<<<wn / 256, 256, 0, stream>>>(Wv, Wvb, wn);
  cvt_f32_bf16<<<wn / 256, 256, 0, stream>>>(Wo, Wob, wn);

  dim3 gg(Mn / 32, Dm / 64), gb(32);
  gemm_kernel<0><<<gg, gb, 0, stream>>>((const void*)q, Wqb, nullptr, Qbf);
  gemm_kernel<1><<<gg, gb, 0, stream>>>((const void*)k, Wkb, keyo, Kbf);
  gemm_kernel<2><<<gg, gb, 0, stream>>>((const void*)v, Wvb, valo, Vtb);

  attn_kernel<<<dim3(Sn / 16, Bn * Hn), 32, 0, stream>>>(Qbf, Kbf, Vtb, Ctx);

  gemm_kernel<3><<<gg, gb, 0, stream>>>((const void*)Ctx, Wob, out, nullptr);
}
